// MGGAT_55198919688626
// MI455X (gfx1250) — compile-verified, matrix operand order checked by reading, not run on hardware
//
#include <hip/hip_runtime.h>
#include <hip/hip_bf16.h>
#include <math.h>

typedef __attribute__((ext_vector_type(16))) __bf16 v16bf;
typedef __attribute__((ext_vector_type(8)))  float  v8f;

#define NU_N 50000
#define NB_N 50000
#define FU_N 256
#define FB_N 256
#define H_N  64
#define L_N  64
#define EU_N 1600000
#define EB_N 1600000
#define BATCH_N 1000000

// ---------------------------------------------------------------- utilities
__global__ __launch_bounds__(256) void zero_f32(float* __restrict__ p, long long n) {
  long long i = (long long)blockIdx.x * blockDim.x + threadIdx.x;
  if (i < n) p[i] = 0.f;
}

// ---------------------------------------------------------------- WMMA GEMM
// C[M x 64] = epi( A[M x K] @ W[K x 64] ),  K compile-time (64 or 256).
// Block: 8 waves, each wave computes 16 rows x 64 cols (4 wmma tiles).
// W is staged in LDS pre-swizzled into exact B-fragment order so each
// fragment is one 32-byte contiguous per-lane load (2x ds_load_b128).
template <int K>
__global__ __launch_bounds__(256) void gemm64_wmma_bf16(
    const float* __restrict__ A, int M,
    const float* __restrict__ W,
    const float* __restrict__ bias1, const float* __restrict__ bias2,
    const float* __restrict__ Cin,  const float* __restrict__ post,
    float* __restrict__ C, int relu)
{
  constexpr int KSTEPS = K / 32;
  // layout: [ntile(4)][kstep][laneslot(32)][elem(16)]  (== K*64 bf16 total)
  __shared__ __align__(32) __bf16 sW[4 * KSTEPS * 32 * 16];

  const int tid = threadIdx.x;
  // ---- stage W into LDS, permuted to B-fragment order -------------------
  for (int idx = tid; idx < K * 64; idx += 256) {
    const int k = idx >> 6, n = idx & 63;
    const int kstep = k >> 5, ko = k & 31;
    const int half = (ko >> 3) & 1;                 // ISA 16-bit B layout
    const int e    = (ko & 7) | ((ko & 16) >> 1);   // element within v16bf
    const int pos  = ((((n >> 4) * KSTEPS + kstep) * 32) + half * 16 + (n & 15)) * 16 + e;
    sW[pos] = (__bf16)W[idx];
  }
  __syncthreads();

  const int wid  = tid >> 5;
  const int lane = tid & 31;
  const int half = lane >> 4;
  const int mr   = lane & 15;
  const int row0 = blockIdx.x * 128 + wid * 16;

  int rowA = row0 + mr; if (rowA >= M) rowA = M - 1;   // clamp; store guarded
  const float* __restrict__ Abase = A + (size_t)rowA * K;

  v8f acc0 = {}, acc1 = {}, acc2 = {}, acc3 = {};
#pragma unroll
  for (int kk = 0; kk < K; kk += 32) {
    if (kk + 32 < K) __builtin_prefetch(Abase + kk + 32, 0, 0);  // global_prefetch_b8
    // A fragment: lane-half owns K {0-7,16-23} or {8-15,24-31} -> two
    // contiguous 8-float runs => 4x global_load_b128 + v_cvt_pk_bf16_f32.
    const float4* Ap0 = (const float4*)(Abase + kk + half * 8);
    const float4* Ap1 = (const float4*)(Abase + kk + 16 + half * 8);
    const float4 x0 = Ap0[0], x1 = Ap0[1], x2 = Ap1[0], x3 = Ap1[1];
    v16bf af;
    af[0]  = (__bf16)x0.x; af[1]  = (__bf16)x0.y; af[2]  = (__bf16)x0.z; af[3]  = (__bf16)x0.w;
    af[4]  = (__bf16)x1.x; af[5]  = (__bf16)x1.y; af[6]  = (__bf16)x1.z; af[7]  = (__bf16)x1.w;
    af[8]  = (__bf16)x2.x; af[9]  = (__bf16)x2.y; af[10] = (__bf16)x2.z; af[11] = (__bf16)x2.w;
    af[12] = (__bf16)x3.x; af[13] = (__bf16)x3.y; af[14] = (__bf16)x3.z; af[15] = (__bf16)x3.w;

    const int kstep = kk >> 5;
    const v16bf b0 = *(const v16bf*)&sW[(((0 * KSTEPS + kstep) * 32) + lane) * 16];
    const v16bf b1 = *(const v16bf*)&sW[(((1 * KSTEPS + kstep) * 32) + lane) * 16];
    const v16bf b2 = *(const v16bf*)&sW[(((2 * KSTEPS + kstep) * 32) + lane) * 16];
    const v16bf b3 = *(const v16bf*)&sW[(((3 * KSTEPS + kstep) * 32) + lane) * 16];
    acc0 = __builtin_amdgcn_wmma_f32_16x16x32_bf16(false, af, false, b0, (short)0, acc0, false, false);
    acc1 = __builtin_amdgcn_wmma_f32_16x16x32_bf16(false, af, false, b1, (short)0, acc1, false, false);
    acc2 = __builtin_amdgcn_wmma_f32_16x16x32_bf16(false, af, false, b2, (short)0, acc2, false, false);
    acc3 = __builtin_amdgcn_wmma_f32_16x16x32_bf16(false, af, false, b3, (short)0, acc3, false, false);
  }

  // ---- epilogue ---------------------------------------------------------
  v8f accs[4] = {acc0, acc1, acc2, acc3};
#pragma unroll
  for (int nt = 0; nt < 4; ++nt) {
    const int col = nt * 16 + mr;
    float badd = 0.f;
    if (bias1) badd += bias1[col];
    if (bias2) badd += bias2[col];
#pragma unroll
    for (int r = 0; r < 8; ++r) {
      const int row = row0 + r + 8 * half;        // C/D: VGPR r -> M = r + 8*half
      if (row < M) {
        const size_t idx = (size_t)row * 64 + col;
        float v = accs[nt][r] + badd;
        if (Cin)  v += Cin[idx];
        if (relu) v = fmaxf(v, 0.f);
        if (post) v += post[idx];
        C[idx] = v;
      }
    }
  }
}

// ------------------------------------------------------- attention pre-scores
__global__ __launch_bounds__(256) void node_scores(
    const float* __restrict__ feat, const float* __restrict__ a,
    float* __restrict__ s_src, float* __restrict__ s_dst, int n)
{
  int i = blockIdx.x * blockDim.x + threadIdx.x;
  if (i >= n) return;
  const float4* row = (const float4*)(feat + (size_t)i * 64);
  float ss = 0.f, sd = 0.f;
#pragma unroll
  for (int j = 0; j < 16; ++j) {
    float4 v = row[j];
    ss += v.x * a[4*j] + v.y * a[4*j+1] + v.z * a[4*j+2] + v.w * a[4*j+3];
    sd += v.x * a[64+4*j] + v.y * a[64+4*j+1] + v.z * a[64+4*j+2] + v.w * a[64+4*j+3];
  }
  s_src[i] = ss;
  s_dst[i] = sd;
}

// -------------------------------------------------- edge weights + denominator
__global__ __launch_bounds__(256) void edge_weights(
    const int* __restrict__ edges, const float* __restrict__ s_src,
    const float* __restrict__ s_dst, const float* __restrict__ graph_w,
    const int* __restrict__ etype, float* __restrict__ w,
    float* __restrict__ denom, int E)
{
  int e = blockIdx.x * blockDim.x + threadIdx.x;
  if (e >= E) return;
  const int s = edges[2 * e], d = edges[2 * e + 1];
  float x = s_src[s] + s_dst[d];
  x = (x > 0.f) ? x : 0.2f * x;                   // leaky_relu(., 0.2)
  if (graph_w) x += graph_w[etype[e]];
  const float wv = 1.f / (1.f + __expf(-x));      // sigmoid
  w[e] = wv;
  atomicAdd(&denom[s], wv);
}

// ------------------------------------------- normalized weighted neighbor sum
// 16 threads per edge, each handles 4 consecutive features (float4 gather).
__global__ __launch_bounds__(256) void edge_aggregate(
    const int* __restrict__ edges, const float* __restrict__ w,
    const float* __restrict__ denom, const float* __restrict__ feat,
    float* __restrict__ out, int E)
{
  long long t = (long long)blockIdx.x * blockDim.x + threadIdx.x;
  if (t >= (long long)E * 16) return;
  const int e = (int)(t >> 4);
  const int q = (int)(t & 15) * 4;
  const int s = edges[2 * e], d = edges[2 * e + 1];
  const float ww = w[e] / fmaxf(denom[s], 1e-8f);
  const float4 f = *(const float4*)(feat + (size_t)d * 64 + q);
  float* op = out + (size_t)s * 64 + q;
  atomicAdd(op + 0, ww * f.x);
  atomicAdd(op + 1, ww * f.y);
  atomicAdd(op + 2, ww * f.z);
  atomicAdd(op + 3, ww * f.w);
}

// -------------------------------------------------------------- final scoring
__global__ __launch_bounds__(256) void score_pairs(
    const int* __restrict__ uidx, const int* __restrict__ bidx,
    const float* __restrict__ uemb, const float* __restrict__ bemb,
    const float* __restrict__ ubias, const float* __restrict__ bbias,
    const float* __restrict__ gbias, float* __restrict__ pred, int n)
{
  int i = blockIdx.x * blockDim.x + threadIdx.x;
  if (i >= n) return;
  const int u = uidx[i], b = bidx[i];
  const float4* ur = (const float4*)(uemb + (size_t)u * 64);
  const float4* br = (const float4*)(bemb + (size_t)b * 64);
  float s = 0.f;
#pragma unroll
  for (int j = 0; j < 16; ++j) {
    float4 a = ur[j], c = br[j];
    s += a.x * c.x + a.y * c.y + a.z * c.z + a.w * c.w;
  }
  s += ubias[u] + bbias[b] + gbias[0];
  pred[i] = 4.f / (1.f + __expf(-s)) + 1.f;
}

// ------------------------------------------------------------------- launcher
extern "C" void kernel_launch(void* const* d_in, const int* in_sizes, int n_in,
                              void* d_out, int out_size, void* d_ws, size_t ws_size,
                              hipStream_t stream) {
  (void)in_sizes; (void)n_in; (void)out_size; (void)ws_size;

  const float* uf    = (const float*)d_in[0];
  const float* bfeat = (const float*)d_in[1];
  const int*   ue    = (const int*)d_in[2];
  const int*   be    = (const int*)d_in[3];
  const int*   betyp = (const int*)d_in[4];
  const int*   uidx  = (const int*)d_in[5];
  const int*   bidx  = (const int*)d_in[6];
  const float* Wu    = (const float*)d_in[7];
  const float* Wb    = (const float*)d_in[8];
  const float* a_u   = (const float*)d_in[9];
  const float* a_b   = (const float*)d_in[10];
  const float* gw    = (const float*)d_in[11];
  const float* Wug   = (const float*)d_in[12];
  const float* bug   = (const float*)d_in[13];
  const float* Wus   = (const float*)d_in[14];
  const float* bus   = (const float*)d_in[15];
  const float* Wbg   = (const float*)d_in[16];
  const float* bbg   = (const float*)d_in[17];
  const float* Wbs   = (const float*)d_in[18];
  const float* bbs   = (const float*)d_in[19];
  const float* Wuo   = (const float*)d_in[20];
  const float* buo   = (const float*)d_in[21];
  const float* Wbo   = (const float*)d_in[22];
  const float* bbo   = (const float*)d_in[23];
  const float* ubase = (const float*)d_in[24];
  const float* bbase = (const float*)d_in[25];
  const float* ubias = (const float*)d_in[26];
  const float* bbias = (const float*)d_in[27];
  const float* gbias = (const float*)d_in[28];

  float* pred = (float*)d_out;
  float* uemb = pred + BATCH_N;
  float* bemb = uemb + (size_t)NU_N * L_N;

  // workspace layout
  float* ws = (float*)d_ws;
  size_t o = 0;
  auto take = [&](size_t n) { float* p = ws + o; o += n; return p; };
  float* u1   = take((size_t)NU_N * 64);
  float* b1   = take((size_t)NB_N * 64);
  float* ssu  = take(NU_N);  float* sdu = take(NU_N);
  float* ssb  = take(NB_N);  float* sdb = take(NB_N);
  float* we_u = take(EU_N);  float* we_b = take(EB_N);
  float* denu = take(NU_N);  float* denb = take(NB_N);
  float* u2   = take((size_t)NU_N * 64);
  float* b2   = take((size_t)NB_N * 64);
  float* tu   = take((size_t)NU_N * 64);
  float* tb   = take((size_t)NB_N * 64);
  float* u3   = take((size_t)NU_N * 64);
  float* b3   = take((size_t)NB_N * 64);

  const int gU = (NU_N + 127) / 128, gB = (NB_N + 127) / 128;

  // Stage 1: u1 = uf@Wu, b1 = bf@Wb
  gemm64_wmma_bf16<256><<<gU, 256, 0, stream>>>(uf, NU_N, Wu, nullptr, nullptr, nullptr, nullptr, u1, 0);
  gemm64_wmma_bf16<256><<<gB, 256, 0, stream>>>(bfeat, NB_N, Wb, nullptr, nullptr, nullptr, nullptr, b1, 0);

  // Stage 2: per-node attention scores
  node_scores<<<(NU_N + 255) / 256, 256, 0, stream>>>(u1, a_u, ssu, sdu, NU_N);
  node_scores<<<(NB_N + 255) / 256, 256, 0, stream>>>(b1, a_b, ssb, sdb, NB_N);

  // Stage 3: zero accumulators
  zero_f32<<<(NU_N + 255) / 256, 256, 0, stream>>>(denu, NU_N);
  zero_f32<<<(NB_N + 255) / 256, 256, 0, stream>>>(denb, NB_N);
  zero_f32<<<((long long)NU_N * 64 + 255) / 256, 256, 0, stream>>>(u2, (long long)NU_N * 64);
  zero_f32<<<((long long)NB_N * 64 + 255) / 256, 256, 0, stream>>>(b2, (long long)NB_N * 64);

  // Stage 4: edge weights + denominators
  edge_weights<<<(EU_N + 255) / 256, 256, 0, stream>>>(ue, ssu, sdu, nullptr, nullptr, we_u, denu, EU_N);
  edge_weights<<<(EB_N + 255) / 256, 256, 0, stream>>>(be, ssb, sdb, gw, betyp, we_b, denb, EB_N);

  // Stage 5: normalized scatter-aggregate (u2, b2)
  edge_aggregate<<<(unsigned)(((long long)EU_N * 16 + 255) / 256), 256, 0, stream>>>(ue, we_u, denu, u1, u2, EU_N);
  edge_aggregate<<<(unsigned)(((long long)EB_N * 16 + 255) / 256), 256, 0, stream>>>(be, we_b, denb, b1, b2, EB_N);

  // Stage 6: skip path  t = feat@Ws + (bs + bg)
  gemm64_wmma_bf16<256><<<gU, 256, 0, stream>>>(uf, NU_N, Wus, bus, bug, nullptr, nullptr, tu, 0);
  gemm64_wmma_bf16<256><<<gB, 256, 0, stream>>>(bfeat, NB_N, Wbs, bbs, bbg, nullptr, nullptr, tb, 0);

  // Stage 7: u3 = relu(u2@Wg + t)
  gemm64_wmma_bf16<64><<<gU, 256, 0, stream>>>(u2, NU_N, Wug, nullptr, nullptr, tu, nullptr, u3, 1);
  gemm64_wmma_bf16<64><<<gB, 256, 0, stream>>>(b2, NB_N, Wbg, nullptr, nullptr, tb, nullptr, b3, 1);

  // Stage 8: emb = relu(u3@Wo + bo) + base  -> straight into d_out
  gemm64_wmma_bf16<64><<<gU, 256, 0, stream>>>(u3, NU_N, Wuo, buo, nullptr, nullptr, ubase, uemb, 1);
  gemm64_wmma_bf16<64><<<gB, 256, 0, stream>>>(b3, NB_N, Wbo, bbo, nullptr, nullptr, bbase, bemb, 1);

  // Stage 9: pair scoring
  score_pairs<<<(BATCH_N + 255) / 256, 256, 0, stream>>>(uidx, bidx, uemb, bemb,
                                                         ubias, bbias, gbias, pred, BATCH_N);
}
